// DebiasSoftConLoss_13322988552200
// MI455X (gfx1250) — compile-verified, matrix operand order checked by reading, not run on hardware
//
#include <hip/hip_runtime.h>
#include <hip/hip_bf16.h>
#include <stdint.h>

typedef __attribute__((ext_vector_type(8)))  _Float16 v8h;
typedef __attribute__((ext_vector_type(16))) _Float16 v16h;
typedef __attribute__((ext_vector_type(8)))  float    v8f;

#define D_DIM 128
#define ROW_BYTES 256                         // D_DIM * sizeof(f16)
#define LDS_ROW_BYTES 272                     // +16B pad -> conflict-free ds_load_b128
#define CHUNK_TILES 8                         // column tiles staged per async chunk
#define CHUNK_ROWS (CHUNK_TILES * 16)         // 128 contrast rows per chunk
#define CHUNK_BYTES (CHUNK_ROWS * LDS_ROW_BYTES) // 34816 B per buffer
#define THREADS 128                           // 4 waves per block
#define XFERS_PER_THREAD ((CHUNK_ROWS * (ROW_BYTES / 16)) / THREADS) // 16
#define INV_T (1.0f / 0.07f)
#define LOG2E 1.4426950408889634f
#define LN2   0.6931471805599453f
#define C_EXP (INV_T * LOG2E)                 // (dot-1)*C_EXP = log2-domain shifted logit
#define EPS_L 1e-9f

// Raw hardware base-2 exp/log (no subnormal range fixup); inputs are bounded.
#if __has_builtin(__builtin_amdgcn_exp2f)
#define FAST_EXP2(x) __builtin_amdgcn_exp2f(x)
#else
extern "C" __device__ float __ocml_native_exp2_f32(float);
#define FAST_EXP2(x) __ocml_native_exp2_f32(x)
#endif
#if __has_builtin(__builtin_amdgcn_logf)
#define FAST_LOG2(x) __builtin_amdgcn_logf(x)
#else
extern "C" __device__ float __ocml_native_log2_f32(float);
#define FAST_LOG2(x) __ocml_native_log2_f32(x)
#endif

// ---------------------------------------------------------------------------
// Kernel 1: features [B, V, D] f32 -> contrast [V*B, D] f16 (view-major)
// ---------------------------------------------------------------------------
__global__ void supcon_convert_f16(const float* __restrict__ feats,
                                   _Float16* __restrict__ contrast,
                                   int B, int V) {
    int idx = blockIdx.x * blockDim.x + threadIdx.x;
    int total = B * V * D_DIM;
    if (idx >= total) return;
    int d   = idx & (D_DIM - 1);
    int row = idx >> 7;
    int v   = row / B;
    int b   = row - v * B;
    contrast[idx] = (_Float16)feats[(size_t)(b * V + v) * D_DIM + d];
}

// ---------------------------------------------------------------------------
// Async-stage one chunk (CHUNK_ROWS rows of f16 contrast) into LDS at bufBase.
// ---------------------------------------------------------------------------
__device__ __forceinline__ void async_load_chunk(const _Float16* contrast, int N,
                                                 uint32_t bufBase, int chunk) {
#pragma unroll
    for (int it = 0; it < XFERS_PER_THREAD; ++it) {
        uint32_t id   = (uint32_t)(it * THREADS) + threadIdx.x;
        uint32_t row  = id >> 4;                  // row within chunk (0..127)
        uint32_t seg  = (id & 15u) << 4;          // 16B segment within row
        uint32_t grow = (uint32_t)chunk * CHUNK_ROWS + row;
        if (grow >= (uint32_t)N) grow = (uint32_t)(N - 1);   // tail clamp
        uint32_t lds = bufBase + row * LDS_ROW_BYTES + seg;
        uint64_t ga  = (uint64_t)(uintptr_t)contrast
                     + (uint64_t)grow * ROW_BYTES + seg;
        asm volatile("global_load_async_to_lds_b128 %0, %1, off"
                     :: "v"(lds), "v"(ga) : "memory");
    }
}

// ---------------------------------------------------------------------------
// Kernel 2: fused SupCon loss. One wave per 16-row tile of the N x N logits.
// ---------------------------------------------------------------------------
__global__ void __launch_bounds__(THREADS)
supcon_loss_kernel(const _Float16* __restrict__ contrast,
                   const float* __restrict__ max_probs,
                   const int* __restrict__ labels,
                   float* __restrict__ out,
                   int B, int V) {
    const int N = B * V;
    const int numTiles  = N >> 4;
    const int numChunks = (numTiles + CHUNK_TILES - 1) / CHUNK_TILES;

    extern __shared__ char smem[];
    int2* s_ml = (int2*)smem;                       // {mp_bits, label} per b
    const uint32_t bufBase0 = (uint32_t)B * 8u;
    const uint32_t bufBase1 = bufBase0 + CHUNK_BYTES;

    for (int i = threadIdx.x; i < B; i += THREADS)
        s_ml[i] = make_int2(__float_as_int(max_probs[i]), labels[i]);

    async_load_chunk(contrast, N, bufBase0, 0);     // kick off chunk 0
    __syncthreads();                                // s_ml visible

    const int lane = threadIdx.x & 31;
    // Force the wave index into an SGPR: rt becomes provably wave-uniform, so
    // the diagonal-tile branch compiles to s_cmp/s_cbranch (EXEC untouched).
    const int wave = __builtin_amdgcn_readfirstlane(threadIdx.x >> 5);
    const int rowTile = blockIdx.x * (THREADS / 32) + wave;
    const bool valid = rowTile < numTiles;
    const int rt = valid ? rowTile : 0;             // clamp; keep barriers uniform

    const int l16 = lane & 15;
    const int hi  = lane >> 4;
    const int rowBase = rt * 16;

    // ---- A fragments: 16 rows x 128 f16, loaded once from global. ----
    const _Float16* arow = contrast + (size_t)(rowBase + l16) * D_DIM;
    v16h afrag[4];
#pragma unroll
    for (int kk = 0; kk < 4; ++kk) {
        const _Float16* p = arow + kk * 32;
        v8h a0 = *(const v8h*)(p + hi * 8);
        v8h a1 = *(const v8h*)(p + 16 + hi * 8);
#pragma unroll
        for (int t = 0; t < 8; ++t) { afrag[kk][t] = a0[t]; afrag[kk][8 + t] = a1[t]; }
    }

    // Per-lane row state: rows rowBase + r + 8*hi. (mp_i only needed at the end)
    float rmp[8]; int rlab[8];
#pragma unroll
    for (int r = 0; r < 8; ++r) {
        int gb = rowBase + r + 8 * hi;
        if (gb >= B) gb -= B; if (gb >= B) gb -= B;     // V is tiny
        int2 ml = s_ml[gb];
        rmp[r]  = __int_as_float(ml.x);
        rlab[r] = ml.y;
    }
    float accL[8], accS0[8], accS1[8];
#pragma unroll
    for (int r = 0; r < 8; ++r) { accL[r] = 0.0f; accS0[r] = 0.0f; accS1[r] = 0.0f; }

    int cb = l16;                                   // column index mod B, incremental

    for (int chunk = 0; chunk < numChunks; ++chunk) {
        const uint32_t curBase = (chunk & 1) ? bufBase1 : bufBase0;
        if (chunk + 1 < numChunks) {
            async_load_chunk(contrast, N,
                             (chunk & 1) ? bufBase0 : bufBase1, chunk + 1);
            asm volatile("s_wait_asynccnt %0" :: "n"(XFERS_PER_THREAD) : "memory");
        } else {
            asm volatile("s_wait_asynccnt 0" ::: "memory");
        }
        __syncthreads();                            // chunk data ready for all waves

        int tmax = numTiles - chunk * CHUNK_TILES;
        if (tmax > CHUNK_TILES) tmax = CHUNK_TILES;

#pragma unroll 2
        for (int t = 0; t < tmax; ++t) {
            const int ct = chunk * CHUNK_TILES + t;
            const char* rb = smem + curBase
                           + (uint32_t)(t * 16 + l16) * LDS_ROW_BYTES
                           + hi * 16;

            v16h bfrag[4];
#pragma unroll
            for (int kk = 0; kk < 4; ++kk) {
                v8h b0 = *(const v8h*)(rb + kk * 64);
                v8h b1 = *(const v8h*)(rb + kk * 64 + 32);
#pragma unroll
                for (int q = 0; q < 8; ++q) { bfrag[kk][q] = b0[q]; bfrag[kk][8 + q] = b1[q]; }
            }

            v8f c = {};
#pragma unroll
            for (int kk = 0; kk < 4; ++kk) {
                c = __builtin_amdgcn_wmma_f32_16x16x32_f16(
                        false, afrag[kk], false, bfrag[kk],
                        (short)0, c, false, false);
            }

            int2 ml = s_ml[cb];                     // one ds_load_b64
            const float cmp = __int_as_float(ml.x); // mp_j (mp_i factored out)
            const int   clb = ml.y;

            if (ct != rt) {                         // scalar-uniform: hot path
#pragma unroll
                for (int r = 0; r < 8; ++r) {
                    float lsh2 = (c[r] - 1.0f) * C_EXP;        // log2-domain logit
                    float e    = FAST_EXP2(lsh2);              // raw v_exp_f32
                    float m    = (clb == rlab[r]) ? cmp : 0.0f;
                    accL[r]  += e;
                    accS0[r] += m;
                    accS1[r]  = fmaf(m, lsh2, accS1[r]);
                }
            } else {                                // diagonal tile (1 of numTiles)
#pragma unroll
                for (int r = 0; r < 8; ++r) {
                    float lsh2 = (c[r] - 1.0f) * C_EXP;
                    float e    = FAST_EXP2(lsh2);
                    float m    = (clb == rlab[r]) ? cmp : 0.0f;
                    if (l16 == r + 8 * hi) { e = 0.0f; m = 0.0f; }  // self-contrast
                    accL[r]  += e;
                    accS0[r] += m;
                    accS1[r]  = fmaf(m, lsh2, accS1[r]);
                }
            }

            cb += 16; if (cb >= B) cb -= B;         // incremental mod B
        }
        __syncthreads();                            // all waves done with curBase
    }

    // ---- Reduce across the 16 lanes of each half-wave (columns). ----
#pragma unroll
    for (int r = 0; r < 8; ++r) {
#pragma unroll
        for (int s = 1; s < 16; s <<= 1) {
            accL[r]  += __shfl_xor(accL[r],  s, 32);
            accS0[r] += __shfl_xor(accS0[r], s, 32);
            accS1[r] += __shfl_xor(accS1[r], s, 32);
        }
    }

    // Lane 0 -> rows rowBase..+7, lane 16 -> rows rowBase+8..+15.
    if (valid && l16 == 0) {
        float local = 0.0f;
#pragma unroll
        for (int r = 0; r < 8; ++r) {
            float L  = accL[r];
            float S0 = accS0[r];                    // Σ same*mp_j (mp_i factored out)
            float S1 = accS1[r];                    // Σ same*mp_j*lsh2 (log2 units)
            if (S0 != 0.0f && rmp[r] != 0.0f) {
                // loss_row = -ln2*(S1 - S0*log2(L+eps))/S0  (mp_i cancels)
                local += -LN2 * (S1 - S0 * FAST_LOG2(L + EPS_L)) / S0;
            }
        }
        atomicAdd(out, local / (float)N);
    }
}

// ---------------------------------------------------------------------------
extern "C" void kernel_launch(void* const* d_in, const int* in_sizes, int n_in,
                              void* d_out, int out_size, void* d_ws, size_t ws_size,
                              hipStream_t stream) {
    const float* feats     = (const float*)d_in[0];
    const float* max_probs = (const float*)d_in[1];
    const int*   labels    = (const int*)d_in[2];
    float*       out       = (float*)d_out;

    const int B = in_sizes[1];                      // 4096
    const int V = in_sizes[0] / (B * D_DIM);        // 2
    const int N = B * V;                            // 8192

    _Float16* contrast = (_Float16*)d_ws;           // N * 128 f16 = 2 MB

    hipMemsetAsync(d_out, 0, (size_t)out_size * sizeof(float), stream);

    {   // 1) f32 -> f16 view-major conversion.
        int total = N * D_DIM;
        int threads = 256;
        int blocks = (total + threads - 1) / threads;
        supcon_convert_f16<<<blocks, threads, 0, stream>>>(feats, contrast, B, V);
    }

    {   // 2) Fused WMMA SupCon loss with async double-buffered LDS staging.
        const int numTiles = N / 16;                // 512
        const int wavesPerBlock = THREADS / 32;     // 4
        const int blocks = (numTiles + wavesPerBlock - 1) / wavesPerBlock; // 128
        const size_t lds = (size_t)B * 8 + 2 * (size_t)CHUNK_BYTES;        // ~100 KB
        supcon_loss_kernel<<<blocks, THREADS, lds, stream>>>(
            contrast, max_probs, labels, out, B, V);
    }
}
